// multigrancnn_29703993819505
// MI455X (gfx1250) — matrix operand order repaired and checked by reading, not compiled, not run on hardware
//
#include <hip/hip_runtime.h>

// ---------------------------------------------------------------------------
// MultiGranCNN forward on MI455X (gfx1250, wave32).
// GEMMs (conv1d-k2 towers, qt = qg@bw, interaction matmul) run on
// v_wmma_f32_16x16x32_bf16 with LDS-staged, double-buffered tiles fed by
// GLOBAL_LOAD_ASYNC_TO_LDS_B128 (ASYNCcnt) when available.
// ---------------------------------------------------------------------------

typedef __attribute__((ext_vector_type(16))) __bf16 v16bf;
typedef __attribute__((ext_vector_type(8)))  float  v8f;
typedef int v4i_ __attribute__((__vector_size__(16)));

struct FragU { uint4 q0, q1; };

__device__ __forceinline__ v8f zero8() {
  v8f z = {0.f,0.f,0.f,0.f,0.f,0.f,0.f,0.f};
  return z;
}

// ---- CDNA5 async global->LDS copy (ASYNCcnt) with graceful fallback -------
#if defined(__has_builtin)
#if __has_builtin(__builtin_amdgcn_global_load_async_to_lds_b128) && \
    __has_builtin(__builtin_amdgcn_s_wait_asynccnt)
#define USE_ASYNC_LDS 1
#endif
#endif
#ifndef USE_ASYNC_LDS
#define USE_ASYNC_LDS 0
#endif

__device__ __forceinline__ void copy16_async(const __bf16* g, __bf16* l) {
#if USE_ASYNC_LDS
  __builtin_amdgcn_global_load_async_to_lds_b128(
      (__attribute__((address_space(1))) v4i_*)g,
      (__attribute__((address_space(3))) v4i_*)l, 0, 0);
#else
  *(uint4*)l = *(const uint4*)g;
#endif
}
__device__ __forceinline__ void wait_async_lds() {
#if USE_ASYNC_LDS
  __builtin_amdgcn_s_wait_asynccnt(0);
#endif
}

__device__ __forceinline__ float sigmoidf_(float x) { return 1.f / (1.f + expf(-x)); }

// ---------------------------------------------------------------------------
// Weight prep
// ---------------------------------------------------------------------------
__global__ void prep_conv_w_kernel(const float* __restrict__ w, __bf16* __restrict__ out,
                                   int C, int Kc) {
  int idx = blockIdx.x * blockDim.x + threadIdx.x;
  int K = 2 * Kc;
  if (idx >= 300 * K) return;
  int k = idx % K, o = idx / K;
  int t = (k >= Kc) ? 1 : 0;
  int c = k - t * Kc;
  float v = (c < C) ? w[((size_t)o * C + c) * 2 + t] : 0.f;
  out[idx] = (__bf16)v;
}

__global__ void prep_bw_kernel(const float* __restrict__ bw, __bf16* __restrict__ out) {
  int idx = blockIdx.x * blockDim.x + threadIdx.x;
  if (idx >= 300 * 320) return;
  int d = idx % 320, e = idx / 320;
  out[idx] = (__bf16)((d < 300) ? bw[(size_t)d * 300 + e] : 0.f);
}

// Padded bf16 copy of layer-1 inputs: (nB, L+8, C), 4 zero rows each side.
__global__ void pad_convert_kernel(const float* __restrict__ src, __bf16* __restrict__ dst,
                                   int nB, int L, int C) {
  size_t idx = (size_t)blockIdx.x * blockDim.x + threadIdx.x;
  size_t total = (size_t)nB * (L + 8) * C;
  if (idx >= total) return;
  int c = (int)(idx % C);
  size_t t = idx / C;
  int lp = (int)(t % (L + 8));
  int b  = (int)(t / (L + 8));
  float v = 0.f;
  if (lp >= 4 && lp < L + 4) v = src[((size_t)b * L + (lp - 4)) * C + c];
  dst[idx] = (__bf16)v;
}

// ---------------------------------------------------------------------------
// Ordered top-k over one row (wave cooperative, wave32).
// ---------------------------------------------------------------------------
template <int EMAX, typename F>
__device__ void kmax_row(const float* __restrict__ src, int n, int stride, int k, F&& wr) {
  const int lane = threadIdx.x & 31;
  const int E = (n + 31) >> 5;
  unsigned keys[EMAX];
#pragma unroll
  for (int e = 0; e < EMAX; ++e) {
    unsigned kk = 0u;
    if (e < E) {
      int idx = e * 32 + lane;
      if (idx < n) {
        unsigned f = __float_as_uint(src[(size_t)idx * stride]);
        kk = f ^ ((unsigned)((int)f >> 31) | 0x80000000u);
      }
    }
    keys[e] = kk;
  }
  unsigned t = 0;
  for (int bit = 31; bit >= 0; --bit) {
    unsigned cand = t | (1u << bit);
    int c = 0;
#pragma unroll
    for (int e = 0; e < EMAX; ++e) {
      if (e >= E) break;
      c += (keys[e] >= cand) ? 1 : 0;
    }
    for (int o = 16; o; o >>= 1) c += __shfl_xor(c, o, 32);
    if (c >= k) t = cand;
  }
  int cgt = 0;
#pragma unroll
  for (int e = 0; e < EMAX; ++e) {
    if (e >= E) break;
    cgt += (keys[e] > t) ? 1 : 0;
  }
  for (int o = 16; o; o >>= 1) cgt += __shfl_xor(cgt, o, 32);
  const int tieQ = k - cgt;
  int totGT = 0, totEQ = 0;
  const unsigned ltm = (1u << lane) - 1u;
  for (int e = 0; e < EMAX; ++e) {
    if (e >= E) break;
    unsigned ky = keys[e];
    bool valid = (e * 32 + lane) < n;
    unsigned bGT = (unsigned)__ballot(valid && (ky > t));
    unsigned bEQ = (unsigned)__ballot(valid && (ky == t));
    int gtB = totGT + __popc(bGT & ltm);
    int eqB = totEQ + __popc(bEQ & ltm);
    int pos = -1;
    if (valid && ky > t)                     pos = gtB + (eqB < tieQ ? eqB : tieQ);
    else if (valid && ky == t && eqB < tieQ) pos = gtB + eqB;
    if (pos >= 0 && pos < k) {
      float v = (ky & 0x80000000u) ? __uint_as_float(ky ^ 0x80000000u)
                                   : __uint_as_float(~ky);
      wr(pos, v);
    }
    totGT += __popc(bGT);
    totEQ += __popc(bEQ);
  }
}

__global__ void kmax_ch_kernel(const float* __restrict__ X, __bf16* __restrict__ Cc,
                               int nRows, int Lsrc, int concatBatchRows) {
  int wid = blockIdx.x * (blockDim.x >> 5) + (threadIdx.x >> 5);
  if (wid >= nRows) return;
  int b = wid / Lsrc, l = wid % Lsrc;
  const float* src = X + (size_t)wid * 1024;
  __bf16* dst = Cc + ((size_t)b * concatBatchRows + l) * 320;
  kmax_row<32>(src, 1024, 1, 300, [&](int pos, float v) { dst[pos] = (__bf16)v; });
}

template <int EMAX>
__global__ void kmax_len_kernel(const float* __restrict__ Y, int Lc, int k,
                                __bf16* __restrict__ P, __bf16* __restrict__ Cc,
                                int concatBatchRows, int concatOff, int nBatch) {
  int wid = blockIdx.x * (blockDim.x >> 5) + (threadIdx.x >> 5);
  if (wid >= nBatch * 300) return;
  int batch = wid / 300, o = wid % 300;
  const float* src = Y + (size_t)batch * Lc * 300 + o;
  __bf16* cdst = Cc + ((size_t)batch * concatBatchRows + concatOff) * 320 + o;
  __bf16* pdst = P ? (P + ((size_t)batch * (k + 8) + 4) * 320 + o) : (__bf16*)nullptr;
  kmax_row<EMAX>(src, Lc, 300, k, [&](int pos, float v) {
    cdst[(size_t)pos * 320] = (__bf16)v;
    if (pdst) pdst[(size_t)pos * 320] = (__bf16)v;
  });
}

// ---------------------------------------------------------------------------
// LDS-staged WMMA GEMM core pieces. Block = 256 threads = 8 waves,
// output tile 128(M) x 64(N); per 32-K step stage A(128x32) + B(64x32) bf16
// via async copies, double buffered; fragments gathered from LDS (ds ops).
// A-fragment per ISA 7.12.2 (lane r,h -> K runs {8h..}, {16+8h..});
// B-fragment same K order as A (lane n,h -> K runs {8h..}, {16+8h..}).
// ---------------------------------------------------------------------------
struct StageCtx {
  const __bf16 *gA0, *gA1, *gB;
  int lA0, lA1, lB;
};

__device__ __forceinline__ void stage_issue(const StageCtx& S, __bf16* sA, __bf16* sB,
                                            int buf, int s) {
  copy16_async(S.gA0 + (size_t)s * 32, sA + buf * 4096 + S.lA0);
  copy16_async(S.gA1 + (size_t)s * 32, sA + buf * 4096 + S.lA1);
  copy16_async(S.gB  + (size_t)s * 32, sB + buf * 2048 + S.lB);
}

__device__ __forceinline__ void gemm_staged_loop(const StageCtx& S, __bf16* sA, __bf16* sB,
                                                 int nsteps, int wave, int r, int h,
                                                 v8f acc[4]) {
  stage_issue(S, sA, sB, 0, 0);
  for (int s = 0; s < nsteps; ++s) {
    const int buf = s & 1;
    wait_async_lds();
    __syncthreads();
    if (s + 1 < nsteps) stage_issue(S, sA, sB, buf ^ 1, s + 1);
    const __bf16* a = sA + buf * 4096 + (wave * 16 + r) * 32;
    union { FragU u; v16bf v; } fa;
    fa.u.q0 = *(const uint4*)(a + 8 * h);
    fa.u.q1 = *(const uint4*)(a + 16 + 8 * h);
#pragma unroll
    for (int i = 0; i < 4; ++i) {
      const __bf16* bp = sB + buf * 2048 + (16 * i + r) * 32;
      union { FragU u; v16bf v; } fb;
      fb.u.q0 = *(const uint4*)(bp + 8 * h);
      fb.u.q1 = *(const uint4*)(bp + 16 + 8 * h);
      acc[i] = __builtin_amdgcn_wmma_f32_16x16x32_bf16(false, fa.v, false, fb.v,
                                                       (short)0, acc[i], false, false);
    }
    __syncthreads();
  }
}

// conv1d-k2 as GEMM: A = padded input (nB, Lc+1, C), row l spans [l*C,(l+2)*C),
// K = 2C; W rows (300, K); Y = tanh(A@W^T + bias) -> (nB, Lc, 300) f32.
__global__ void __launch_bounds__(256)
gemm_conv_kernel(const __bf16* __restrict__ A, const __bf16* __restrict__ W,
                 const float* __restrict__ bias, float* __restrict__ Y,
                 int C, int Lc) {
  __shared__ __bf16 sA[2 * 128 * 32];
  __shared__ __bf16 sB[2 * 64 * 32];
  const int K = 2 * C;
  const int tid = threadIdx.x, lane = tid & 31, wave = tid >> 5;
  const int m0 = blockIdx.x * 128, n0 = blockIdx.y * 64, batch = blockIdx.z;
  const int r = lane & 15, h = lane >> 4;

  StageCtx S;
  {
    int c0 = tid, c1 = tid + 256;
    int rA0 = c0 >> 2, cA0 = (c0 & 3) * 8;
    int rA1 = c1 >> 2, cA1 = (c1 & 3) * 8;
    int g0 = m0 + rA0; if (g0 > Lc - 1) g0 = Lc - 1;
    int g1 = m0 + rA1; if (g1 > Lc - 1) g1 = Lc - 1;
    S.gA0 = A + ((size_t)batch * (Lc + 1) + g0) * C + cA0;
    S.gA1 = A + ((size_t)batch * (Lc + 1) + g1) * C + cA1;
    S.lA0 = rA0 * 32 + cA0;
    S.lA1 = rA1 * 32 + cA1;
    int rB = tid >> 2, cB = (tid & 3) * 8;
    int gn = n0 + rB; if (gn > 299) gn = 299;
    S.gB = W + (size_t)gn * K + cB;
    S.lB = rB * 32 + cB;
  }
  v8f acc[4];
#pragma unroll
  for (int i = 0; i < 4; ++i) acc[i] = zero8();
  gemm_staged_loop(S, sA, sB, K >> 5, wave, r, h, acc);
#pragma unroll
  for (int i = 0; i < 4; ++i) {
    int n = n0 + 16 * i + r;
    if (n < 300) {
      float bs = bias[n];
#pragma unroll
      for (int p = 0; p < 8; ++p) {
        int m = m0 + wave * 16 + p + 8 * h;
        if (m < Lc) Y[((size_t)batch * Lc + m) * 300 + n] = tanhf(acc[i][p] + bs);
      }
    }
  }
}

// qt = qg @ bw : A (7680 x 320) bf16, B = bwT (300 x 320), out bf16 (7680 x 320)
__global__ void __launch_bounds__(256)
gemm_qt_kernel(const __bf16* __restrict__ qg, const __bf16* __restrict__ Wbw,
               __bf16* __restrict__ qt) {
  __shared__ __bf16 sA[2 * 128 * 32];
  __shared__ __bf16 sB[2 * 64 * 32];
  const int tid = threadIdx.x, lane = tid & 31, wave = tid >> 5;
  const int m0 = blockIdx.x * 128, n0 = blockIdx.y * 64;
  const int r = lane & 15, h = lane >> 4;
  StageCtx S;
  {
    int c0 = tid, c1 = tid + 256;
    int rA0 = c0 >> 2, cA0 = (c0 & 3) * 8;
    int rA1 = c1 >> 2, cA1 = (c1 & 3) * 8;
    S.gA0 = qg + (size_t)(m0 + rA0) * 320 + cA0;
    S.gA1 = qg + (size_t)(m0 + rA1) * 320 + cA1;
    S.lA0 = rA0 * 32 + cA0;
    S.lA1 = rA1 * 32 + cA1;
    int rB = tid >> 2, cB = (tid & 3) * 8;
    int gn = n0 + rB; if (gn > 299) gn = 299;
    S.gB = Wbw + (size_t)gn * 320 + cB;
    S.lB = rB * 32 + cB;
  }
  v8f acc[4];
#pragma unroll
  for (int i = 0; i < 4; ++i) acc[i] = zero8();
  gemm_staged_loop(S, sA, sB, 10, wave, r, h, acc);
#pragma unroll
  for (int i = 0; i < 4; ++i) {
    int n = n0 + 16 * i + r;
#pragma unroll
    for (int p = 0; p < 8; ++p) {
      int m = m0 + wave * 16 + p + 8 * h;
      float v = (n < 300) ? acc[i][p] : 0.f;  // zero K-pad columns
      qt[(size_t)m * 320 + n] = (__bf16)v;
    }
  }
}

// M[sb] = sigmoid(qt[sb%16] (480x320) @ dgB[sb] (960x320)^T + bb)
__global__ void __launch_bounds__(256)
gemm_interact_kernel(const __bf16* __restrict__ qt, const __bf16* __restrict__ dgB,
                     const float* __restrict__ bb, float* __restrict__ M) {
  __shared__ __bf16 sA[2 * 128 * 32];
  __shared__ __bf16 sB[2 * 64 * 32];
  const int sb = blockIdx.z, b = sb & 15;
  const int tid = threadIdx.x, lane = tid & 31, wave = tid >> 5;
  const int m0 = blockIdx.x * 128, n0 = blockIdx.y * 64;
  const int r = lane & 15, h = lane >> 4;
  StageCtx S;
  {
    int c0 = tid, c1 = tid + 256;
    int rA0 = c0 >> 2, cA0 = (c0 & 3) * 8;
    int rA1 = c1 >> 2, cA1 = (c1 & 3) * 8;
    int g0 = m0 + rA0; if (g0 > 479) g0 = 479;
    int g1 = m0 + rA1; if (g1 > 479) g1 = 479;
    S.gA0 = qt + ((size_t)b * 480 + g0) * 320 + cA0;
    S.gA1 = qt + ((size_t)b * 480 + g1) * 320 + cA1;
    S.lA0 = rA0 * 32 + cA0;
    S.lA1 = rA1 * 32 + cA1;
    int rB = tid >> 2, cB = (tid & 3) * 8;
    S.gB = dgB + ((size_t)sb * 960 + (n0 + rB)) * 320 + cB;
    S.lB = rB * 32 + cB;
  }
  v8f acc[4];
#pragma unroll
  for (int i = 0; i < 4; ++i) acc[i] = zero8();
  gemm_staged_loop(S, sA, sB, 10, wave, r, h, acc);
  const float bias0 = bb[0];
#pragma unroll
  for (int i = 0; i < 4; ++i) {
    int n = n0 + 16 * i + r;
#pragma unroll
    for (int p = 0; p < 8; ++p) {
      int m = m0 + wave * 16 + p + 8 * h;
      if (m < 480) M[((size_t)sb * 480 + m) * 960 + n] = sigmoidf_(acc[i][p] + bias0);
    }
  }
}

// ---------------------------------------------------------------------------
__global__ void pool_M_kernel(const float* __restrict__ M, float* __restrict__ out) {
  int idx = blockIdx.x * blockDim.x + threadIdx.x;
  if (idx >= 80 * 40 * 40) return;
  int j = idx % 40, t = idx / 40, i = t % 40, n = t / 40;
  const float* base = M + ((size_t)n * 480 + i * 12) * 960 + j * 24;
  float m = -3.4e38f;
  for (int a = 0; a < 12; ++a)
    for (int c = 0; c < 24; ++c) m = fmaxf(m, base[a * 960 + c]);
  out[idx] = m;
}

__device__ __forceinline__ float xp_ld(const float* x, int r, int s) {
  return (r >= 0 && r < 40 && s >= 0 && s < 40) ? x[r * 40 + s] : 0.f;
}
__global__ void conv2d_block_kernel(const float* __restrict__ X, const float* __restrict__ W,
                                    const float* __restrict__ Bv, float* __restrict__ Out) {
  int idx = blockIdx.x * blockDim.x + threadIdx.x;
  if (idx >= 80 * 40 * 40) return;
  int j = idx % 40, t = idx / 40, i = t % 40, n = t / 40;
  const float* x = X + (size_t)n * 1600;
  int si = (i * 47) / 40, ei = ((i + 1) * 47 + 39) / 40;
  int sj = (j * 47) / 40, ej = ((j + 1) * 47 + 39) / 40;
  float m = -3.4e38f;
  for (int ii = si; ii < ei; ++ii) {
    for (int jj = sj; jj < ej; ++jj) {
      float x00 = xp_ld(x, ii - 4, jj - 4);
      float x01 = xp_ld(x, ii - 4, jj - 3);
      float x10 = xp_ld(x, ii - 3, jj - 4);
      float x11 = xp_ld(x, ii - 3, jj - 3);
      for (int c = 0; c < 50; ++c) {
        float v = Bv[c] + x00 * W[c * 4 + 0] + x01 * W[c * 4 + 1]
                        + x10 * W[c * 4 + 2] + x11 * W[c * 4 + 3];
        m = fmaxf(m, sigmoidf_(v));
      }
    }
  }
  Out[idx] = m;
}

__global__ void final_kernel(const float* __restrict__ feat, const float* __restrict__ lw,
                             const float* __restrict__ lb, float* __restrict__ out) {
  int n = blockIdx.x, lane = threadIdx.x;
  float acc = 0.f;
  for (int i = lane; i < 40; i += 32) {
    const float* fr = feat + ((size_t)n * 40 + i) * 40;
    float d = 0.f;
    for (int jj = 0; jj < 40; ++jj) d += fr[jj] * lw[jj];
    acc += sigmoidf_(d + lb[0]) * lw[i];
  }
  for (int o = 16; o; o >>= 1) acc += __shfl_xor(acc, o, 32);
  if (lane == 0) out[n] = sigmoidf_(acc + lb[0]);
}

// ---------------------------------------------------------------------------
extern "C" void kernel_launch(void* const* d_in, const int* in_sizes, int n_in,
                              void* d_out, int out_size, void* d_ws, size_t ws_size,
                              hipStream_t stream) {
  const float* query   = (const float*)d_in[0];
  const float* pos_doc = (const float*)d_in[1];
  const float* neg_docs= (const float*)d_in[2];
  const float* qw1=(const float*)d_in[3],  *qb1=(const float*)d_in[4];
  const float* qw2=(const float*)d_in[5],  *qb2=(const float*)d_in[6];
  const float* qw3=(const float*)d_in[7],  *qb3=(const float*)d_in[8];
  const float* dw1=(const float*)d_in[9],  *db1=(const float*)d_in[10];
  const float* dw2=(const float*)d_in[11], *db2=(const float*)d_in[12];
  const float* dw3=(const float*)d_in[13], *db3=(const float*)d_in[14];
  const float* bw =(const float*)d_in[15], *bb =(const float*)d_in[16];
  const float* mw1=(const float*)d_in[17], *mb1=(const float*)d_in[18];
  const float* mw2=(const float*)d_in[19], *mb2=(const float*)d_in[20];
  const float* mw3=(const float*)d_in[21], *mb3=(const float*)d_in[22];
  const float* lw =(const float*)d_in[23], *lb =(const float*)d_in[24];
  float* out = (float*)d_out;

  const int Bq=16, ND=80, Lq=239, Ld=479, C1=1024;
  const int kq1=160, kq2=80, kd1=320, kd2=160;
  const int LcQ1=Lq+7, LcD1=Ld+7;            // 246, 486
  const int LcQ2=kq1+7, LcD2=kd1+7;          // 167, 327
  const int LcQ3=kq2+7, LcD3=kd2+7;          // 87, 167

  size_t off = 0;
  auto alloc = [&](size_t bytes) -> void* {
    void* p = (char*)d_ws + off;
    off = (off + bytes + 255) & ~(size_t)255;
    return p;
  };
  __bf16* Wq1 = (__bf16*)alloc((size_t)300*2048*2);
  __bf16* Wd1 = (__bf16*)alloc((size_t)300*2048*2);
  __bf16* Wq2 = (__bf16*)alloc((size_t)300*640*2);
  __bf16* Wq3 = (__bf16*)alloc((size_t)300*640*2);
  __bf16* Wd2 = (__bf16*)alloc((size_t)300*640*2);
  __bf16* Wd3 = (__bf16*)alloc((size_t)300*640*2);
  __bf16* Wbw = (__bf16*)alloc((size_t)300*320*2);
  __bf16* padQ1 = (__bf16*)alloc((size_t)Bq*(Lq+8)*C1*2);
  __bf16* padD1 = (__bf16*)alloc((size_t)ND*(Ld+8)*C1*2);
  size_t szPadQ2 = (size_t)Bq*(kq1+8)*320*2;  __bf16* padQ2 = (__bf16*)alloc(szPadQ2);
  size_t szPadQ3 = (size_t)Bq*(kq2+8)*320*2;  __bf16* padQ3 = (__bf16*)alloc(szPadQ3);
  size_t szPadD2 = (size_t)ND*(kd1+8)*320*2;  __bf16* padD2 = (__bf16*)alloc(szPadD2);
  size_t szPadD3 = (size_t)ND*(kd2+8)*320*2;  __bf16* padD3 = (__bf16*)alloc(szPadD3);
  size_t szQg  = (size_t)Bq*480*320*2;        __bf16* qg  = (__bf16*)alloc(szQg);
  size_t szDgB = (size_t)ND*960*320*2;        __bf16* dgB = (__bf16*)alloc(szDgB);
  __bf16* qt   = (__bf16*)alloc((size_t)Bq*480*320*2);
  float* Yq1 = (float*)alloc((size_t)Bq*LcQ1*300*4);
  float* Yd1 = (float*)alloc((size_t)ND*LcD1*300*4);
  float* Yq2 = (float*)alloc((size_t)Bq*LcQ2*300*4);
  float* Yd2 = (float*)alloc((size_t)ND*LcD2*300*4);
  float* Yq3 = (float*)alloc((size_t)Bq*LcQ3*300*4);
  float* Yd3 = (float*)alloc((size_t)ND*LcD3*300*4);
  float* Mbuf = (float*)alloc((size_t)ND*480*960*4);
  float* pool0 = (float*)alloc((size_t)ND*1600*4);
  float* cbuf  = (float*)alloc((size_t)ND*1600*4);

  auto cdiv = [](long a, long b) { return (int)((a + b - 1) / b); };

  // --- weight prep ---
  prep_conv_w_kernel<<<cdiv(300*2048,256),256,0,stream>>>(qw1, Wq1, 1024, 1024);
  prep_conv_w_kernel<<<cdiv(300*2048,256),256,0,stream>>>(dw1, Wd1, 1024, 1024);
  prep_conv_w_kernel<<<cdiv(300*640,256),256,0,stream>>>(qw2, Wq2, 300, 320);
  prep_conv_w_kernel<<<cdiv(300*640,256),256,0,stream>>>(qw3, Wq3, 300, 320);
  prep_conv_w_kernel<<<cdiv(300*640,256),256,0,stream>>>(dw2, Wd2, 300, 320);
  prep_conv_w_kernel<<<cdiv(300*640,256),256,0,stream>>>(dw3, Wd3, 300, 320);
  prep_bw_kernel<<<cdiv(300*320,256),256,0,stream>>>(bw, Wbw);

  // --- zero padded/concat buffers (capturable memset nodes) ---
  (void)hipMemsetAsync(padQ2, 0, szPadQ2, stream);
  (void)hipMemsetAsync(padQ3, 0, szPadQ3, stream);
  (void)hipMemsetAsync(padD2, 0, szPadD2, stream);
  (void)hipMemsetAsync(padD3, 0, szPadD3, stream);
  (void)hipMemsetAsync(qg,  0, szQg,  stream);
  (void)hipMemsetAsync(dgB, 0, szDgB, stream);

  // --- layer-1 padded bf16 inputs ---
  pad_convert_kernel<<<cdiv((long)Bq*(Lq+8)*C1,256),256,0,stream>>>(query, padQ1, Bq, Lq, C1);
  pad_convert_kernel<<<cdiv((long)16*(Ld+8)*C1,256),256,0,stream>>>(pos_doc, padD1, 16, Ld, C1);
  pad_convert_kernel<<<cdiv((long)64*(Ld+8)*C1,256),256,0,stream>>>(
      neg_docs, padD1 + (size_t)16*(Ld+8)*C1, 64, Ld, C1);

  // --- g0: ordered top-300 channels ---
  kmax_ch_kernel<<<cdiv(Bq*Lq,4),128,0,stream>>>(query, qg, Bq*Lq, Lq, 480);
  kmax_ch_kernel<<<cdiv(16*Ld,4),128,0,stream>>>(pos_doc, dgB, 16*Ld, Ld, 960);
  kmax_ch_kernel<<<cdiv(64*Ld,4),128,0,stream>>>(neg_docs, dgB + (size_t)16*960*320, 64*Ld, Ld, 960);

  // --- tower layer 1 (WMMA GEMM + tanh), then ordered top-k over length ---
  gemm_conv_kernel<<<dim3(cdiv(LcQ1,128),5,Bq),256,0,stream>>>(padQ1, Wq1, qb1, Yq1, 1024, LcQ1);
  gemm_conv_kernel<<<dim3(cdiv(LcD1,128),5,ND),256,0,stream>>>(padD1, Wd1, db1, Yd1, 1024, LcD1);
  kmax_len_kernel<8><<<cdiv(Bq*300,4),128,0,stream>>>(Yq1, LcQ1, kq1, padQ2, qg, 480, 239, Bq);
  kmax_len_kernel<16><<<cdiv(ND*300,4),128,0,stream>>>(Yd1, LcD1, kd1, padD2, dgB, 960, 479, ND);

  // --- tower layer 2 ---
  gemm_conv_kernel<<<dim3(cdiv(LcQ2,128),5,Bq),256,0,stream>>>(padQ2, Wq2, qb2, Yq2, 320, LcQ2);
  gemm_conv_kernel<<<dim3(cdiv(LcD2,128),5,ND),256,0,stream>>>(padD2, Wd2, db2, Yd2, 320, LcD2);
  kmax_len_kernel<6><<<cdiv(Bq*300,4),128,0,stream>>>(Yq2, LcQ2, kq2, padQ3, qg, 480, 239+kq1, Bq);
  kmax_len_kernel<11><<<cdiv(ND*300,4),128,0,stream>>>(Yd2, LcD2, kd2, padD3, dgB, 960, 479+kd1, ND);

  // --- tower layer 3 (k = 1, concat only) ---
  gemm_conv_kernel<<<dim3(cdiv(LcQ3,128),5,Bq),256,0,stream>>>(padQ3, Wq3, qb3, Yq3, 320, LcQ3);
  gemm_conv_kernel<<<dim3(cdiv(LcD3,128),5,ND),256,0,stream>>>(padD3, Wd3, db3, Yd3, 320, LcD3);
  kmax_len_kernel<3><<<cdiv(Bq*300,4),128,0,stream>>>(Yq3, LcQ3, 1, (__bf16*)nullptr, qg, 480, 479, Bq);
  kmax_len_kernel<6><<<cdiv(ND*300,4),128,0,stream>>>(Yd3, LcD3, 1, (__bf16*)nullptr, dgB, 960, 959, ND);

  // --- qt = qg @ bw (WMMA), then interaction M = sigmoid(qt . dg^T + bb) ---
  gemm_qt_kernel<<<dim3(60,5),256,0,stream>>>(qg, Wbw, qt);
  gemm_interact_kernel<<<dim3(4,15,ND),256,0,stream>>>(qt, dgB, bb, Mbuf);

  // --- pooling + conv2d tower + head ---
  pool_M_kernel<<<cdiv(ND*1600,256),256,0,stream>>>(Mbuf, pool0);
  conv2d_block_kernel<<<cdiv(ND*1600,256),256,0,stream>>>(pool0, mw1, mb1, cbuf);
  conv2d_block_kernel<<<cdiv(ND*1600,256),256,0,stream>>>(cbuf, mw2, mb2, pool0);
  conv2d_block_kernel<<<cdiv(ND*1600,256),256,0,stream>>>(pool0, mw3, mb3, cbuf);
  final_kernel<<<80,32,0,stream>>>(cbuf, lw, lb, out);

  (void)in_sizes; (void)n_in; (void)out_size; (void)ws_size;
}
